// ContTimeLSTM_62045097558492
// MI455X (gfx1250) — compile-verified, matrix operand order checked by reading, not run on hardware
//
#include <hip/hip_runtime.h>
#include <hip/hip_bf16.h>

// Continuous-time LSTM (NHP) for MI455X (gfx1250, wave32, WMMA).
// Phase A: x_proj = x @ Wx^T  (16384x512x3584 bf16 WMMA GEMM, fp32 accum)
// Phase B: persistent 32-workgroup recurrence kernel; per step:
//          h = o*tanh(decay(c)) -> grid barrier -> stage h into LDS ->
//          WMMA h@Wh^T (A and B fragments both from LDS) -> gate/state update
//          (states live in registers across all 512 steps).

typedef __attribute__((ext_vector_type(16))) __bf16 v16bf;
typedef __attribute__((ext_vector_type(8)))  float  v8f;

#define Bz 32
#define Lz 512
#define Iz 512
#define Dz 512
#define NO 3584      // 7*D
#define KW 1024      // I + D   (weight row stride)
#define NWG 32       // workgroups in recurrence kernel (one per 16-col slab)

static __device__ __forceinline__ __bf16 f2bf(float f) {
  unsigned u = __builtin_bit_cast(unsigned, f);
  unsigned r = u + 0x7fffu + ((u >> 16) & 1u);     // round-to-nearest-even
  unsigned short s = (unsigned short)(r >> 16);
  return __builtin_bit_cast(__bf16, s);
}

static __device__ __forceinline__ v8f wmma_bf16(v16bf a, v16bf b, v8f c) {
  return __builtin_amdgcn_wmma_f32_16x16x32_bf16(false, a, false, b, (short)0, c,
                                                 false, false);
}

static __device__ __forceinline__ float sigmoidf_(float x) {
  return 1.0f / (1.0f + __expf(-x));
}
static __device__ __forceinline__ float softplusf_(float x) {
  return fmaxf(x, 0.0f) + log1pf(__expf(-fabsf(x)));
}

// ---------------------------------------------------------------- converts
__global__ __launch_bounds__(256) void k_cvt_x(const float* __restrict__ x,
                                               __bf16* __restrict__ xb) {
  size_t i = (size_t)blockIdx.x * 256 + threadIdx.x;   // exact multiple
  xb[i] = f2bf(x[i]);
}

__global__ __launch_bounds__(256) void k_cvt_w(const float* __restrict__ w,
                                               __bf16* __restrict__ wxb,
                                               __bf16* __restrict__ whb) {
  size_t i = (size_t)blockIdx.x * 256 + threadIdx.x;   // over NO*Iz, exact
  size_t n = i / Iz, k = i % Iz;
  wxb[i] = f2bf(w[n * KW + k]);          // x-part  weight[:, :I]
  whb[i] = f2bf(w[n * KW + Iz + k]);     // h-part  weight[:, I:]
}

// ---------------------------------------------------------------- Phase A GEMM
// C(16384x3584) = A(16384x512) * B(3584x512)^T ; bf16 in, f32 out.
// Block = 8 waves covering 128(M) x 64(N); wave = 2x2 tiles of 16x16.
__global__ __launch_bounds__(256) void k_xproj(const __bf16* __restrict__ xb,
                                               const __bf16* __restrict__ wxb,
                                               float* __restrict__ xp) {
  const int lane = threadIdx.x & 31;
  const int wid  = threadIdx.x >> 5;
  const int lr = lane & 15, hi = lane >> 4;
  const int wm = wid & 3, wn = wid >> 2;
  const int mb = blockIdx.y * 128 + wm * 32;
  const int nb = blockIdx.x * 64  + wn * 32;

  v8f acc00 = {}, acc01 = {}, acc10 = {}, acc11 = {};
  for (int ko = 0; ko < Iz; ko += 32) {
    const int kb = ko + hi * 16;
    v16bf a0 = *(const v16bf*)(xb  + (size_t)(mb + lr)      * Iz + kb);
    v16bf a1 = *(const v16bf*)(xb  + (size_t)(mb + 16 + lr) * Iz + kb);
    v16bf b0 = *(const v16bf*)(wxb + (size_t)(nb + lr)      * Iz + kb);
    v16bf b1 = *(const v16bf*)(wxb + (size_t)(nb + 16 + lr) * Iz + kb);
    acc00 = wmma_bf16(a0, b0, acc00);
    acc01 = wmma_bf16(a0, b1, acc01);
    acc10 = wmma_bf16(a1, b0, acc10);
    acc11 = wmma_bf16(a1, b1, acc11);
  }
  // C layout: VGPR v -> row (v + 8*hi), col = lane%16 within the 16x16 tile.
  #pragma unroll
  for (int v = 0; v < 8; ++v) {
    int r0 = mb + v + 8 * hi, r1 = r0 + 16;
    int c0 = nb + lr,         c1 = c0 + 16;
    xp[(size_t)r0 * NO + c0] = acc00[v];
    xp[(size_t)r0 * NO + c1] = acc01[v];
    xp[(size_t)r1 * NO + c0] = acc10[v];
    xp[(size_t)r1 * NO + c1] = acc11[v];
  }
}

// ---------------------------------------------------------------- grid barrier
static __device__ __forceinline__ void grid_barrier(unsigned* cnt, unsigned* gen) {
  __syncthreads();
  if (threadIdx.x == 0) {
    unsigned g = __hip_atomic_load(gen, __ATOMIC_ACQUIRE, __HIP_MEMORY_SCOPE_AGENT);
    unsigned a = __hip_atomic_fetch_add(cnt, 1u, __ATOMIC_ACQ_REL,
                                        __HIP_MEMORY_SCOPE_AGENT);
    if (a == NWG - 1) {
      __hip_atomic_store(cnt, 0u, __ATOMIC_RELAXED, __HIP_MEMORY_SCOPE_AGENT);
      __hip_atomic_fetch_add(gen, 1u, __ATOMIC_ACQ_REL, __HIP_MEMORY_SCOPE_AGENT);
    } else {
      while (__hip_atomic_load(gen, __ATOMIC_ACQUIRE, __HIP_MEMORY_SCOPE_AGENT) == g)
        __builtin_amdgcn_s_sleep(2);
    }
  }
  __syncthreads();
}

// ---------------------------------------------------------------- Phase B
// 32 blocks x 256 threads. Block wg owns output columns [wg*16, wg*16+16).
// LDS: Wh slab 112x512 bf16 (114688 B) + h stage 32x512 bf16 (32768 B)
//      + proj exchange 7x32x16 f32 (14336 B)  = 161792 B  (< 320 KB WGP).
#define LDSW_B (112 * 512 * 2)
#define LDSH_B (32 * 512 * 2)
__global__ __launch_bounds__(256) void k_recur(
    const float* __restrict__ xp, const float* __restrict__ td,
    const float* __restrict__ st0, const float* __restrict__ bias,
    const __bf16* __restrict__ whb, __bf16* __restrict__ hbuf,
    float* __restrict__ out_h, float* __restrict__ out_s,
    unsigned* __restrict__ sync) {
  extern __shared__ __attribute__((aligned(128))) char smem[];
  __bf16* ldsW = (__bf16*)smem;                         // 112*512 bf16
  __bf16* ldsH = (__bf16*)(smem + LDSW_B);              // 32*512 bf16
  float*  ldsP = (float*)(smem + LDSW_B + LDSH_B);      // 7*32*16 f32

  const int tid  = threadIdx.x;
  const int wg   = blockIdx.x;
  const int j0   = wg * 16;
  const int lane = tid & 31, wid = tid >> 5;
  const int lr = lane & 15, hi = lane >> 4;

  // Stage Wh slab once: slab row r = g*16+jc  <-  weight row g*512 + j0 + jc.
  for (int idx = tid; idx < 112 * 64; idx += 256) {     // 64 uint4 per row
    int r = idx >> 6, w4 = idx & 63;
    int g = r >> 4, jc = r & 15;
    const uint4* src = (const uint4*)(whb + (size_t)(g * Dz + j0 + jc) * Dz) + w4;
    ((uint4*)(ldsW + (size_t)r * Dz))[w4] = *src;
  }
  __syncthreads();

  // Each thread owns 2 (batch, col) state elements across all 512 steps.
  const int b0 = tid >> 4,          jc0 = tid & 15;          // batches 0..15
  const int b1 = (tid + 256) >> 4,  jc1 = tid & 15;          // batches 16..31
  const int jj0 = j0 + jc0, jj1 = j0 + jc1;
  float cs0 = st0[b0 * 4 * Dz + 0 * Dz + jj0];
  float ce0 = st0[b0 * 4 * Dz + 1 * Dz + jj0];
  float dc0 = st0[b0 * 4 * Dz + 2 * Dz + jj0];
  float og0 = st0[b0 * 4 * Dz + 3 * Dz + jj0];
  float cs1 = st0[b1 * 4 * Dz + 0 * Dz + jj1];
  float ce1 = st0[b1 * 4 * Dz + 1 * Dz + jj1];
  float dc1 = st0[b1 * 4 * Dz + 2 * Dz + jj1];
  float og1 = st0[b1 * 4 * Dz + 3 * Dz + jj1];

  const float bv = (wid < 7) ? bias[wid * Dz + j0 + lr] : 0.0f;

  float c0 = 0.0f, c1 = 0.0f;
  for (int t = 0; t < Lz; ++t) {
    __bf16* hb = hbuf + (size_t)(t & 1) * (Bz * Dz);

    // ---- phase 1: decay cell, emit h (depends only on carried state + dt_t)
    {
      float dt = td[b0 * Lz + t];
      c0 = ce0 + (cs0 - ce0) * __expf(-dc0 * dt);
      float h = og0 * tanhf(c0);
      out_h[((size_t)b0 * Lz + t) * Dz + jj0] = h;
      hb[b0 * Dz + jj0] = f2bf(h);
    }
    {
      float dt = td[b1 * Lz + t];
      c1 = ce1 + (cs1 - ce1) * __expf(-dc1 * dt);
      float h = og1 * tanhf(c1);
      out_h[((size_t)b1 * Lz + t) * Dz + jj1] = h;
      hb[b1 * Dz + jj1] = f2bf(h);
    }
    // warm next step's x_proj lines while we wait at the barrier
    {
      int tn = (t + 1 < Lz) ? t + 1 : t;
      __builtin_prefetch(
          (const void*)(xp + ((size_t)b0 * Lz + tn) * NO + (tid % 7) * Dz + jj0),
          0, 1);
    }

    grid_barrier(sync, sync + 1);   // h(t) now visible device-wide

    // ---- phase 2a: stage full h(t) (32x512 bf16 = 32KB) into LDS once,
    //      instead of every GEMM wave re-reading it from L2 (7x traffic).
    {
      const uint4* src = (const uint4*)hb;        // 2048 uint4
      uint4* dst = (uint4*)ldsH;
      #pragma unroll
      for (int i = 0; i < 8; ++i) dst[tid + 256 * i] = src[tid + 256 * i];
    }
    __syncthreads();

    // ---- phase 2b: proj = x_proj + bias + h @ Wh^T (waves 0..6, gate = wid)
    if (wid < 7) {
      v8f acc0 = {}, acc1 = {};
      const __bf16* wrow = ldsW + (size_t)(wid * 16 + lr) * Dz;
      const __bf16* h0p  = ldsH + (size_t)lr * Dz;
      const __bf16* h1p  = ldsH + (size_t)(16 + lr) * Dz;
      for (int ko = 0; ko < Dz; ko += 32) {
        const int kb = ko + hi * 16;
        v16bf bf  = *(const v16bf*)(wrow + kb);   // ds_load_b128 x2
        v16bf af0 = *(const v16bf*)(h0p + kb);
        v16bf af1 = *(const v16bf*)(h1p + kb);
        acc0 = wmma_bf16(af0, bf, acc0);
        acc1 = wmma_bf16(af1, bf, acc1);
      }
      const int col = wid * Dz + j0 + lr;
      #pragma unroll
      for (int v = 0; v < 8; ++v) {
        int r0 = v + 8 * hi;        // batch 0..15
        int r1 = 16 + r0;           // batch 16..31
        float p0 = acc0[v] + bv + xp[((size_t)r0 * Lz + t) * NO + col];
        float p1 = acc1[v] + bv + xp[((size_t)r1 * Lz + t) * NO + col];
        ldsP[(wid * 32 + r0) * 16 + lr] = p0;
        ldsP[(wid * 32 + r1) * 16 + lr] = p1;
      }
    }
    __syncthreads();

    // ---- phase 3: gates + state update (registers carry to next step)
    {
      float p[7];
      #pragma unroll
      for (int g = 0; g < 7; ++g) p[g] = ldsP[(g * 32 + b0) * 16 + jc0];
      float ig = sigmoidf_(p[0]), fg = sigmoidf_(p[1]), ieg = sigmoidf_(p[2]);
      float feg = sigmoidf_(p[3]), og = sigmoidf_(p[4]);
      float z = tanhf(p[5]);
      float dn = softplusf_(p[6]);
      float csn = fg * c0 + ig * z;
      float cen = feg * ce0 + ieg * z;
      size_t so = ((size_t)b0 * Lz + t) * 4 * Dz + jj0;
      out_s[so] = csn; out_s[so + Dz] = cen;
      out_s[so + 2 * Dz] = dn; out_s[so + 3 * Dz] = og;
      cs0 = csn; ce0 = cen; dc0 = dn; og0 = og;
    }
    {
      float p[7];
      #pragma unroll
      for (int g = 0; g < 7; ++g) p[g] = ldsP[(g * 32 + b1) * 16 + jc1];
      float ig = sigmoidf_(p[0]), fg = sigmoidf_(p[1]), ieg = sigmoidf_(p[2]);
      float feg = sigmoidf_(p[3]), og = sigmoidf_(p[4]);
      float z = tanhf(p[5]);
      float dn = softplusf_(p[6]);
      float csn = fg * c1 + ig * z;
      float cen = feg * ce1 + ieg * z;
      size_t so = ((size_t)b1 * Lz + t) * 4 * Dz + jj1;
      out_s[so] = csn; out_s[so + Dz] = cen;
      out_s[so + 2 * Dz] = dn; out_s[so + 3 * Dz] = og;
      cs1 = csn; ce1 = cen; dc1 = dn; og1 = og;
    }
    // next iteration's ldsH/ldsP writes happen after grid_barrier's
    // __syncthreads, so phase-3 readers are protected without another barrier.
  }
}

// ---------------------------------------------------------------- launch
extern "C" void kernel_launch(void* const* d_in, const int* in_sizes, int n_in,
                              void* d_out, int out_size, void* d_ws, size_t ws_size,
                              hipStream_t stream) {
  (void)in_sizes; (void)n_in; (void)out_size; (void)ws_size;
  const float* x    = (const float*)d_in[0];   // (B,L,I)
  const float* td   = (const float*)d_in[1];   // (B,L)
  const float* st0  = (const float*)d_in[2];   // (B,4D)
  const float* w    = (const float*)d_in[3];   // (7D, I+D)
  const float* bias = (const float*)d_in[4];   // (7D)
  float* out_h = (float*)d_out;                        // (B,L,D)
  float* out_s = out_h + (size_t)Bz * Lz * Dz;         // (B,L,4D)

  // workspace layout (~259 MB): sync | xb | Wxb | Whb | hbuf(x2) | x_proj
  char* ws = (char*)d_ws;
  size_t off = 0;
  unsigned* sync = (unsigned*)ws;                 off += 256;
  __bf16* xb   = (__bf16*)(ws + off);             off += (size_t)Bz * Lz * Iz * 2;
  __bf16* wxb  = (__bf16*)(ws + off);             off += (size_t)NO * Iz * 2;
  __bf16* whb  = (__bf16*)(ws + off);             off += (size_t)NO * Dz * 2;
  __bf16* hbuf = (__bf16*)(ws + off);             off += 2 * (size_t)Bz * Dz * 2;
  off = (off + 255) & ~(size_t)255;
  float* xp    = (float*)(ws + off);

  hipMemsetAsync(sync, 0, 256, stream);

  k_cvt_x<<<(Bz * Lz * Iz) / 256, 256, 0, stream>>>(x, xb);
  k_cvt_w<<<(NO * Iz) / 256, 256, 0, stream>>>(w, wxb, whb);

  dim3 g1(NO / 64, (Bz * Lz) / 128);
  k_xproj<<<g1, 256, 0, stream>>>(xb, wxb, xp);

  const size_t lds = LDSW_B + LDSH_B + 7 * 32 * 16 * 4;   // 161,792 B
  hipFuncSetAttribute((const void*)k_recur,
                      hipFuncAttributeMaxDynamicSharedMemorySize, (int)lds);
  k_recur<<<NWG, 256, lds, stream>>>(xp, td, st0, bias, whb, hbuf,
                                     out_h, out_s, sync);
}
